// Model_60782377173229
// MI455X (gfx1250) — matrix operand order repaired and checked by reading, not compiled, not run on hardware
//
#include <hip/hip_runtime.h>

typedef unsigned short u16;
typedef __bf16 bf16_t;
typedef bf16_t bf16x16 __attribute__((ext_vector_type(16)));
typedef float f32x8 __attribute__((ext_vector_type(8)));
typedef unsigned uint32x4 __attribute__((ext_vector_type(4)));
typedef int int32x4 __attribute__((ext_vector_type(4)));
typedef int int32x8 __attribute__((ext_vector_type(8)));

#define LL 512
#define NMODE 32

// Tensor Data Mover availability (device pass only; bridge-doc confirmed builtins)
#if defined(__HIP_DEVICE_COMPILE__) && __has_builtin(__builtin_amdgcn_tensor_load_to_lds) && \
    __has_builtin(__builtin_amdgcn_s_wait_tensorcnt)
#define TDM_OK 1
#else
#define TDM_OK 0
#endif

__device__ __forceinline__ u16 f2bf(float x) {
  unsigned u = __float_as_uint(x);
  u += 0x7FFFu + ((u >> 16) & 1u);
  return (u16)(u >> 16);
}
__device__ __forceinline__ float gelu_f(float x) {
  return 0.5f * x * (1.0f + erff(x * 0.70710678118654752f));
}

// ---------------------------------------------------------------------------
// WMMA GEMM: C[M,N] = A[M,K] (bf16; or A[K,M] if AT) @ Bw[K,N] (+bias)(gelu)
// Block tile 128x64, BK=32, 8 waves (4x2), each wave 32x32 = 2x2 WMMA tiles.
// A tiles staged by the Tensor Data Mover (wave 0 issues, TENSORcnt waits);
// B tiles staged with vectorized b128 loads + LDS transpose scatter.
// ---------------------------------------------------------------------------
#define BM 128
#define BN 64
#define BK 32

__device__ __forceinline__ bf16x16 load_fragA(const u16* rowbase, int lane) {
  // 16-bit A 16x32: lanes 0-15 -> k chunks {0..7,16..23}; lanes 16-31 -> {8..15,24..31}
  int r = lane & 15;
  int ko = (lane >> 4) * 8;
  union { bf16x16 v; uint4 q[2]; } u;
  u.q[0] = *(const uint4*)(rowbase + r * BK + ko);
  u.q[1] = *(const uint4*)(rowbase + r * BK + ko + 16);
  return u.v;
}
__device__ __forceinline__ bf16x16 load_fragB(const u16* rowbase, int lane) {
  // B 32x16 (stored N-major in LDS): lanes 0-15 -> K {0..7,16..23}, lanes 16-31 -> K {8..15,24..31}
  int n = lane & 15;
  int ko = (lane >> 4) * 8;
  union { bf16x16 v; uint4 q[2]; } u;
  u.q[0] = *(const uint4*)(rowbase + n * BK + ko);
  u.q[1] = *(const uint4*)(rowbase + n * BK + ko + 16);
  return u.v;
}

// AT: 0 = A row-major [M,K], 1 = A transposed-source [K,M]
// EPI: 0 = none, 1 = +bias, 2 = gelu (no bias)
// OUTBF: 0 = f32 C, 1 = bf16 C
template<int AT, int EPI, int OUTBF>
__global__ __launch_bounds__(256)
void gemm_wmma(const u16* __restrict__ Aptr, const u16* __restrict__ Bw,
               const float* __restrict__ bias, void* __restrict__ Cptr,
               int M, int N, int K, long strideA, long strideC)
{
  __shared__ __align__(16) u16 As[2][BM][BK];
  __shared__ __align__(16) u16 Bs[2][BN][BK];
  const int tid  = threadIdx.x;
  const int lane = tid & 31;
  const int w    = tid >> 5;
  const int wm   = (w & 3) * 32;
  const int wn   = (w >> 2) * 32;
  const int bm0  = blockIdx.y * BM;
  const int bn0  = blockIdx.x * BN;
  const long bz  = blockIdx.z;

  const u16* A0 = Aptr + bz * strideA;
  float* Cf = (float*)Cptr + (OUTBF == 0 ? bz * strideC : 0);
  u16*   Cb = (u16*)Cptr   + (OUTBF == 1 ? bz * strideC : 0);

  auto stageA = [&](int kk, int buf) {
    if (AT == 1) {                       // transpose-stage from A[k][m], lda = M
      #pragma unroll
      for (int i = 0; i < 2; ++i) {      // 512 x 16B chunks, scatter-transpose
        int t2 = tid + i * 256;
        int kr = t2 >> 4;                // 0..31
        int m0 = (t2 & 15) * 8;          // 0..120
        union { uint4 q; u16 e[8]; } u;
        u.q = *(const uint4*)(A0 + (long)(kk + kr) * M + bm0 + m0);
        #pragma unroll
        for (int j = 0; j < 8; ++j) As[buf][m0 + j][kr] = u.e[j];
      }
    } else {
#if TDM_OK
      if (w == 0) {                      // one TDM DMA per block (EXEC-independent)
        unsigned long long ga = (unsigned long long)(A0 + (long)bm0 * K + kk);
        unsigned la = (unsigned)(unsigned long long)&As[buf][0][0];
        uint32x4 g0 = { 1u,                                  // count=1, user D#
                        la,                                  // lds_addr
                        (unsigned)ga,                        // global_addr[31:0]
                        (unsigned)((ga >> 32) & 0x1FFFFFFull) | 0x80000000u }; // addr[56:32] | type=2
        int32x8 g1 = { 0x00010000,                                   // data_size = 2B
                       (int)(((unsigned)K & 0xFFFFu) << 16),         // tensor_dim0 lo16
                       (int)((((unsigned)K >> 16) & 0xFFFFu) |
                             (((unsigned)M & 0xFFFFu) << 16)),       // dim0 hi | dim1 lo
                       (int)((((unsigned)M >> 16) & 0xFFFFu) |
                             ((unsigned)BK << 16)),                  // dim1 hi | tile_dim0
                       BM,                                           // tile_dim1 (tile_dim2=0)
                       K,                                            // tensor_dim0_stride lo32
                       0, 0 };
        int32x4 gz = { 0, 0, 0, 0 };
#if __clang_major__ >= 23
        int32x8 gz8 = { 0, 0, 0, 0, 0, 0, 0, 0 };
        __builtin_amdgcn_tensor_load_to_lds(g0, g1, gz, gz, gz8, 0);
#else
        __builtin_amdgcn_tensor_load_to_lds(g0, g1, gz, gz, 0);
#endif
      }
#else
      #pragma unroll
      for (int i = 0; i < 2; ++i) {      // fallback: direct 16B copies
        int f = tid + i * 256;
        int r = f >> 2;
        int c = (f & 3) * 8;
        *(uint4*)&As[buf][r][c] = *(const uint4*)(A0 + (long)(bm0 + r) * K + kk + c);
      }
#endif
    }
  };
  auto stageB = [&](int kk, int buf) {   // one b128 load + 8-way LDS scatter/thread
    int kr = tid >> 3;                   // 0..31
    int n0 = (tid & 7) * 8;              // 0..56
    union { uint4 q; u16 e[8]; } u;
    u.q = *(const uint4*)(Bw + (long)(kk + kr) * N + bn0 + n0);
    #pragma unroll
    for (int j = 0; j < 8; ++j) Bs[buf][n0 + j][kr] = u.e[j];
  };
  auto tdm_wait = [&]() {
#if TDM_OK
    if (AT == 0 && w == 0) __builtin_amdgcn_s_wait_tensorcnt(0);
#endif
  };

  f32x8 acc[2][2] = {};
  const int nk = K / BK;
  stageA(0, 0);
  stageB(0, 0);
  tdm_wait();
  __syncthreads();
  for (int t = 0; t < nk; ++t) {
    int buf = t & 1;
    if (t + 1 < nk) {
      stageA((t + 1) * BK, buf ^ 1);
      stageB((t + 1) * BK, buf ^ 1);
    }
    if (t + 2 < nk) __builtin_prefetch(Bw + (long)(t + 2) * BK * N + bn0, 0, 1);
    bf16x16 af0 = load_fragA(&As[buf][wm][0], lane);
    bf16x16 af1 = load_fragA(&As[buf][wm + 16][0], lane);
    bf16x16 bf0 = load_fragB(&Bs[buf][wn][0], lane);
    bf16x16 bf1 = load_fragB(&Bs[buf][wn + 16][0], lane);
    acc[0][0] = __builtin_amdgcn_wmma_f32_16x16x32_bf16(false, af0, false, bf0, (short)0, acc[0][0], false, false);
    acc[0][1] = __builtin_amdgcn_wmma_f32_16x16x32_bf16(false, af0, false, bf1, (short)0, acc[0][1], false, false);
    acc[1][0] = __builtin_amdgcn_wmma_f32_16x16x32_bf16(false, af1, false, bf0, (short)0, acc[1][0], false, false);
    acc[1][1] = __builtin_amdgcn_wmma_f32_16x16x32_bf16(false, af1, false, bf1, (short)0, acc[1][1], false, false);
    if (t + 1 < nk) tdm_wait();
    __syncthreads();
  }

  const int rb = (lane >> 4) * 8;   // C layout: lanes 16-31 hold M = v + 8
  const int cn = lane & 15;
  #pragma unroll
  for (int i = 0; i < 2; ++i)
    #pragma unroll
    for (int j = 0; j < 2; ++j) {
      int col = bn0 + wn + j * 16 + cn;
      float bv = (EPI == 1) ? bias[col] : 0.0f;
      #pragma unroll
      for (int v = 0; v < 8; ++v) {
        int row = bm0 + wm + i * 16 + rb + v;
        float val = acc[i][j][v] + bv;
        if (EPI == 2) val = gelu_f(val);
        if (OUTBF == 0) Cf[(long)row * N + col] = val;
        else            Cb[(long)row * N + col] = f2bf(val);
      }
    }
}

// ---------------------------------------------------------------------------
// Replicate np.random.RandomState(0).shuffle(arange(256))[:32] sorted (MT19937)
// ---------------------------------------------------------------------------
__global__ void init_idx_kernel(int* __restrict__ idxout) {
  if (threadIdx.x != 0 || blockIdx.x != 0) return;
  unsigned mt[624];
  mt[0] = 0u;
  for (int i = 1; i < 624; ++i)
    mt[i] = 1812433253u * (mt[i - 1] ^ (mt[i - 1] >> 30)) + (unsigned)i;
  int mti = 624;
  auto next = [&]() -> unsigned {
    if (mti >= 624) {
      for (int i = 0; i < 624; ++i) {
        unsigned y = (mt[i] & 0x80000000u) | (mt[(i + 1) % 624] & 0x7fffffffu);
        mt[i] = mt[(i + 397) % 624] ^ (y >> 1) ^ ((y & 1u) ? 0x9908b0dfu : 0u);
      }
      mti = 0;
    }
    unsigned y = mt[mti++];
    y ^= y >> 11; y ^= (y << 7) & 0x9d2c5680u; y ^= (y << 15) & 0xefc60000u; y ^= y >> 18;
    return y;
  };
  int arr[256];
  for (int i = 0; i < 256; ++i) arr[i] = i;
  for (int i = 255; i > 0; --i) {
    unsigned mask = (unsigned)i;
    mask |= mask >> 1; mask |= mask >> 2; mask |= mask >> 4; mask |= mask >> 8; mask |= mask >> 16;
    unsigned j;
    do { j = next() & mask; } while (j > (unsigned)i);
    int tmp = arr[i]; arr[i] = arr[j]; arr[j] = tmp;
  }
  int sel[NMODE];
  for (int i = 0; i < NMODE; ++i) sel[i] = arr[i];
  for (int i = 1; i < NMODE; ++i) {
    int v = sel[i], j = i - 1;
    while (j >= 0 && sel[j] > v) { sel[j + 1] = sel[j]; --j; }
    sel[j + 1] = v;
  }
  for (int i = 0; i < NMODE; ++i) idxout[i] = sel[i];
}

// DFT basis F[L, 2M] (cos, -sin at selected freqs) and irfft synthesis IB[2M, L]
__global__ void basis_kernel(const int* __restrict__ idx, u16* __restrict__ Fb, u16* __restrict__ IB) {
  int t = blockIdx.x * blockDim.x + threadIdx.x;
  if (t >= LL * NMODE) return;
  int l = t >> 5;
  int m = t & 31;
  {
    int ph = (idx[m] * l) & (LL - 1);
    float ang = 6.2831853071795864f * (float)ph / (float)LL;
    Fb[l * (2 * NMODE) + 2 * m]     = f2bf(__cosf(ang));
    Fb[l * (2 * NMODE) + 2 * m + 1] = f2bf(-__sinf(ang));
  }
  {
    int k = m;                       // ordinal bins 0..31 (torch quirk)
    int ph = (k * l) & (LL - 1);
    float ang = 6.2831853071795864f * (float)ph / (float)LL;
    float sc = (k == 0) ? (1.0f / LL) : (2.0f / LL);
    IB[(2 * k) * LL + l]     = f2bf(sc * __cosf(ang));
    IB[(2 * k + 1) * LL + l] = f2bf((k == 0) ? 0.0f : (-sc * __sinf(ang)));
  }
}

__global__ void cvt_bf16_kernel(const float* __restrict__ src, u16* __restrict__ dst, long n) {
  long t = (long)blockIdx.x * blockDim.x + threadIdx.x;
  if (t < n) dst[t] = f2bf(src[t]);
}

// circular conv1d(k=3) token embedding + sinusoidal positional embedding
// writes f32 (residual stream) and bf16 shadow (GEMM A input)
__global__ void embed_kernel(const float* __restrict__ xe, const float* __restrict__ tw,
                             float* __restrict__ X, u16* __restrict__ Xb) {
  int bl = blockIdx.x;
  int b = bl >> 9;
  int l = bl & 511;
  __shared__ float xs[3][21];
  int tid = threadIdx.x;
  if (tid < 63) {
    int kk = tid / 21, ci = tid % 21;
    int ll = (l + kk - 1 + 512) & 511;
    xs[kk][ci] = xe[((long)b * 512 + ll) * 21 + ci];
  }
  __syncthreads();
  for (int dd = tid; dd < 512; dd += 256) {
    float s = 0.0f;
    #pragma unroll
    for (int kk = 0; kk < 3; ++kk)
      for (int ci = 0; ci < 21; ++ci)
        s += xs[kk][ci] * tw[(kk * 21 + ci) * 512 + dd];
    int i2 = dd >> 1;
    float div = __expf((float)(2 * i2) * (-9.210340371976184f / 512.0f));
    float ang = (float)l * div;
    s += (dd & 1) ? __cosf(ang) : __sinf(ang);
    long o = ((long)b * 512 + l) * 512 + dd;
    X[o] = s;
    Xb[o] = f2bf(s);
  }
}

// per-mode complex mix: og[b,h,o,m] = sum_e xg[b,h,e,m] * (wr + i wi)[h,e,o,m]
__global__ void modemix_kernel(const float* __restrict__ G, const float* __restrict__ fwr,
                               const float* __restrict__ fwi, u16* __restrict__ ogc, int li) {
  int blk = blockIdx.x;
  int m = blk & 31;
  int h = (blk >> 5) & 7;
  int b = blk >> 8;
  int o = threadIdx.x;               // 0..63
  const float* gb = G + (long)b * 512 * 64;
  long wbase = (long)(li * 8 + h) * 131072 + (long)o * 32 + m;
  float ar = 0.f, ai = 0.f;
  for (int e = 0; e < 64; ++e) {
    float xr = gb[(h * 64 + e) * 64 + 2 * m];
    float xi = gb[(h * 64 + e) * 64 + 2 * m + 1];
    float wr = fwr[wbase + (long)e * 2048];
    float wi = fwi[wbase + (long)e * 2048];
    ar += xr * wr - xi * wi;
    ai += xr * wi + xi * wr;
  }
  long orow = (long)b * 512 + h * 64 + o;
  ogc[orow * 64 + 2 * m]     = f2bf(ar);
  ogc[orow * 64 + 2 * m + 1] = f2bf(ai);
}

// out = s - movavg25(s), s = xin + add ; edge-replicated window.
// Each thread handles one (b, d, 128-long L-segment); 4x parallelism vs full column.
__global__ void decomp_kernel(const float* __restrict__ xin, const float* __restrict__ add,
                              float* __restrict__ out, u16* __restrict__ outb) {
  int t = blockIdx.x * blockDim.x + threadIdx.x;   // ((b,seg),d): 32*4*512
  if (t >= 32 * 4 * 512) return;
  int d = t & 511;
  int seg = (t >> 9) & 3;
  int b = t >> 11;
  const float* xb = xin + (long)b * 262144 + d;
  const float* ab = add + (long)b * 262144 + d;
  float* ob = out + (long)b * 262144 + d;
  u16* o16 = outb + (long)b * 262144 + d;
  auto S = [&](int j) {
    j = j < 0 ? 0 : (j > 511 ? 511 : j);
    return xb[(long)j * 512] + ab[(long)j * 512];
  };
  int s0 = seg * 128;
  float sum = 0.0f;
  for (int j = s0 - 12; j <= s0 + 12; ++j) sum += S(j);
  for (int l = s0; l < s0 + 128; ++l) {
    float v = S(l) - sum * (1.0f / 25.0f);
    ob[(long)l * 512] = v;
    o16[(long)l * 512] = f2bf(v);
    sum += S(l + 13) - S(l - 12);
  }
}

__global__ void ln_kernel(const float* __restrict__ X, const float* __restrict__ lw,
                          const float* __restrict__ lb, float* __restrict__ Z) {
  int row = blockIdx.x;
  const float* xr = X + (long)row * 512;
  __shared__ float red[256];
  int tid = threadIdx.x;
  float s = 0.f;
  for (int d = tid; d < 512; d += 256) s += xr[d];
  red[tid] = s; __syncthreads();
  for (int st = 128; st > 0; st >>= 1) { if (tid < st) red[tid] += red[tid + st]; __syncthreads(); }
  float mu = red[0] / 512.0f; __syncthreads();
  float v = 0.f;
  for (int d = tid; d < 512; d += 256) { float u = xr[d] - mu; v += u * u; }
  red[tid] = v; __syncthreads();
  for (int st = 128; st > 0; st >>= 1) { if (tid < st) red[tid] += red[tid + st]; __syncthreads(); }
  float inv = rsqrtf(red[0] / 512.0f + 1e-5f);
  for (int d = tid; d < 512; d += 256)
    Z[(long)row * 512 + d] = (xr[d] - mu) * inv * lw[d] + lb[d];
}

__global__ void colmean_kernel(const float* __restrict__ Z, float* __restrict__ meanL) {
  int t = blockIdx.x * blockDim.x + threadIdx.x;   // (b,d)
  if (t >= 32 * 512) return;
  int b = t >> 9, d = t & 511;
  float s = 0.f;
  for (int l = 0; l < 512; ++l) s += Z[((long)b * 512 + l) * 512 + d];
  meanL[t] = s / 512.0f;
}

__global__ void proj_kernel(const float* __restrict__ Z, const float* __restrict__ meanL,
                            const float* __restrict__ pw, const float* __restrict__ pb,
                            float* __restrict__ out) {
  int b = blockIdx.x;
  int tid = threadIdx.x;
  float acc[10] = {};
  for (int k = tid; k < 512 * 512; k += 256) {
    int d = k & 511;
    float g = gelu_f(Z[(long)b * 262144 + k] - meanL[b * 512 + d]);
    const float* w = pw + (long)k * 10;
    #pragma unroll
    for (int n = 0; n < 10; ++n) acc[n] += g * w[n];
  }
  __shared__ float red[10][256];
  for (int n = 0; n < 10; ++n) red[n][tid] = acc[n];
  __syncthreads();
  for (int st = 128; st > 0; st >>= 1) {
    if (tid < st) for (int n = 0; n < 10; ++n) red[n][tid] += red[n][tid + st];
    __syncthreads();
  }
  if (tid < 10) out[b * 10 + tid] = red[tid][0] + pb[tid];
}

// ---------------------------------------------------------------------------
extern "C" void kernel_launch(void* const* d_in, const int* in_sizes, int n_in,
                              void* d_out, int out_size, void* d_ws, size_t ws_size,
                              hipStream_t stream) {
  (void)in_sizes; (void)n_in; (void)out_size; (void)ws_size;
  const float* x_enc   = (const float*)d_in[0];
  const float* token_w = (const float*)d_in[1];
  const float* Wq      = (const float*)d_in[2];
  const float* bq      = (const float*)d_in[3];
  // d_in[4..7] = Wk,bk,Wv,bv: computed-but-unused in the reference -> skipped
  const float* Wo      = (const float*)d_in[8];
  const float* bo      = (const float*)d_in[9];
  const float* fwr     = (const float*)d_in[10];
  const float* fwi     = (const float*)d_in[11];
  const float* W1      = (const float*)d_in[12];
  const float* W2      = (const float*)d_in[13];
  const float* ln_w    = (const float*)d_in[14];
  const float* ln_b    = (const float*)d_in[15];
  const float* proj_w  = (const float*)d_in[16];
  const float* proj_b  = (const float*)d_in[17];
  float* out = (float*)d_out;

  char* ws = (char*)d_ws;
  auto alloc = [&](size_t bytes) { char* p = ws; ws += (bytes + 255) & ~(size_t)255; return p; };
  float* X0  = (float*)alloc(33554432);           // [B*L, D] f32 residual stream
  float* X1  = (float*)alloc(33554432);
  float* Yb  = (float*)alloc(33554432);
  u16*   X0b = (u16*)alloc(16777216);             // bf16 shadows for GEMM A
  u16*   X1b = (u16*)alloc(16777216);
  u16*   Q   = (u16*)alloc(16777216);             // [B*L, D] bf16
  float* G   = (float*)alloc(4194304);            // [B, he, 64] f32 (re/im)
  u16*   OGC = (u16*)alloc(2097152);              // [B*512, 64] bf16
  u16*   O   = (u16*)alloc(16777216);             // [B*512, 512] bf16
  u16*   Hb  = (u16*)alloc(67108864);             // [B*L, DFF] bf16
  u16*   Wqb = (u16*)alloc(3ul * 262144 * 2);
  u16*   Wob = (u16*)alloc(3ul * 262144 * 2);
  u16*   W1b = (u16*)alloc(3ul * 1048576 * 2);
  u16*   W2b = (u16*)alloc(3ul * 1048576 * 2);
  u16*   Fb  = (u16*)alloc(512 * 64 * 2);
  u16*   IB  = (u16*)alloc(64 * 512 * 2);
  int*   idx = (int*)alloc(256);
  float* meanL = (float*)alloc(32 * 512 * 4);
  float* Z = (float*)Hb;                          // reuse FFN buffer after last layer

  init_idx_kernel<<<1, 1, 0, stream>>>(idx);
  basis_kernel<<<(512 * 32 + 255) / 256, 256, 0, stream>>>(idx, Fb, IB);
  cvt_bf16_kernel<<<(786432 + 255) / 256, 256, 0, stream>>>(Wq, Wqb, 786432);
  cvt_bf16_kernel<<<(786432 + 255) / 256, 256, 0, stream>>>(Wo, Wob, 786432);
  cvt_bf16_kernel<<<(3145728 + 255) / 256, 256, 0, stream>>>(W1, W1b, 3145728);
  cvt_bf16_kernel<<<(3145728 + 255) / 256, 256, 0, stream>>>(W2, W2b, 3145728);
  embed_kernel<<<32 * 512, 256, 0, stream>>>(x_enc, token_w, X0, X0b);

  for (int li = 0; li < 3; ++li) {
    const u16* wq = Wqb + (size_t)li * 262144;
    const u16* wo = Wob + (size_t)li * 262144;
    const u16* w1 = W1b + (size_t)li * 1048576;
    const u16* w2 = W2b + (size_t)li * 1048576;
    // Q = X @ Wq + bq                      (16384 x 512 x 512)
    gemm_wmma<0,1,1><<<dim3(8,128,1),256,0,stream>>>(X0b, wq, bq + li*512, Q, 16384,512,512, 0, 0);
    // G[b] = Q[b]^T @ F                    (512 x 64 x 512, batched over B)
    gemm_wmma<1,0,0><<<dim3(1,4,32),256,0,stream>>>(Q, Fb, nullptr, G, 512,64,512, 262144L, 32768L);
    // per-mode complex 64x64 mix
    modemix_kernel<<<8192, 64, 0, stream>>>(G, fwr, fwi, OGC, li);
    // O = OGC @ IB  (irfft of dense 32-bin prefix)   (16384 x 512 x 64)
    gemm_wmma<0,0,1><<<dim3(8,128,1),256,0,stream>>>(OGC, IB, nullptr, O, 16384,512,64, 0, 0);
    // Y = O(viewed [B,L,D]) @ Wo + bo      (16384 x 512 x 512)
    gemm_wmma<0,1,0><<<dim3(8,128,1),256,0,stream>>>(O, wo, bo + li*512, Yb, 16384,512,512, 0, 0);
    decomp_kernel<<<256, 256, 0, stream>>>(X0, Yb, X1, X1b);
    // H = gelu(X @ W1)                     (16384 x 2048 x 512)
    gemm_wmma<0,2,1><<<dim3(32,128,1),256,0,stream>>>(X1b, w1, nullptr, Hb, 16384,2048,512, 0, 0);
    // Y = H @ W2                           (16384 x 512 x 2048)
    gemm_wmma<0,0,0><<<dim3(8,128,1),256,0,stream>>>(Hb, w2, nullptr, Yb, 16384,512,2048, 0, 0);
    decomp_kernel<<<256, 256, 0, stream>>>(X1, Yb, X0, X0b);
  }

  ln_kernel<<<16384, 256, 0, stream>>>(X0, ln_w, ln_b, Z);
  colmean_kernel<<<64, 256, 0, stream>>>(Z, meanL);
  proj_kernel<<<32, 256, 0, stream>>>(Z, meanL, proj_w, proj_b, out);
}
